// L3KSparseDotProd_71485435674741
// MI455X (gfx1250) — matrix operand order repaired and checked by reading, not compiled, not run on hardware
//
#include <hip/hip_runtime.h>
#include <hip/hip_bf16.h>
#include <stdint.h>

// ---------------------------------------------------------------------------
// Problem constants (from reference setup_inputs)
// ---------------------------------------------------------------------------
#define S_LEN  2048
#define DHEAD  128
#define NKH    4
#define NQH    4
#define NH     16      // NKH * NQH
#define BM     16      // query rows per block
#define SINKN  16
#define NEGV   (-10000.0f)
#define SCALE  0.08838834764831845f  // 1/sqrt(128), folded into Q prep

// workspace layout (bytes)
#define QTOT   (NH  * S_LEN * DHEAD)   // 4,194,304 elements
#define KTOT   (NKH * S_LEN * DHEAD)   // 1,048,576 elements
#define OFF_QB 0u
#define OFF_KT (OFF_QB + (size_t)QTOT * 2)           //  8,388,608
#define OFF_VB (OFF_KT + (size_t)KTOT * 2)           // 10,485,760
#define OFF_QS (OFF_VB + (size_t)KTOT * 2)           // 12,582,912
#define OFF_KS (OFF_QS + (size_t)NH  * S_LEN * 16)   // 13,107,200
// total: 13,238,272 bytes of d_ws used

typedef __attribute__((ext_vector_type(16))) __bf16 v16bf;
typedef __attribute__((ext_vector_type(8)))  float  v8f;

union FragB { v16bf v; uint4 q[2]; unsigned u[8]; };

__device__ __forceinline__ unsigned bf16rn(float x) {
  unsigned u = __float_as_uint(x);
  return (u + 0x7FFFu + ((u >> 16) & 1u)) >> 16;   // round-to-nearest-even
}
__device__ __forceinline__ unsigned pkbf(float lo, float hi) {
  return bf16rn(lo) | (bf16rn(hi) << 16);
}

// ---------------------------------------------------------------------------
// Prep kernels
// ---------------------------------------------------------------------------
__global__ void l3k_cvt_bf16(const float* __restrict__ in,
                             unsigned short* __restrict__ out, int n, float scale) {
  int i = blockIdx.x * blockDim.x + threadIdx.x;
  if (i < n) out[i] = (unsigned short)bf16rn(in[i] * scale);
}

// K [KH][S][D] f32 -> kT [KH][S][D] bf16 (so WMMA B-fragments load contiguously)
__global__ void l3k_cvt_kT(const float* __restrict__ k,
                           unsigned short* __restrict__ kT) {
  int i = blockIdx.x * blockDim.x + threadIdx.x;
  if (i >= KTOT) return;
  int d = i % DHEAD;
  int l = (i / DHEAD) % S_LEN;
  int h = i / (DHEAD * S_LEN);
  (void)d; (void)l; (void)h;
  kT[i] = (unsigned short)bf16rn(k[i]);
}

// pack sign bits: one thread per (row, 32-float word); rows of length 128
__global__ void l3k_sigpack(const float* __restrict__ x,
                            unsigned* __restrict__ sig, int nrows) {
  int i = blockIdx.x * blockDim.x + threadIdx.x;
  if (i >= nrows * 4) return;
  int w = i & 3;
  int row = i >> 2;
  const float* p = x + (size_t)row * DHEAD + w * 32;
  unsigned m = 0u;
#pragma unroll
  for (int j = 0; j < 32; ++j)
    if (p[j] >= 0.0f) m |= (1u << j);
  sig[i] = m;
}

// ---------------------------------------------------------------------------
// Main attention kernel: grid = (S/16, KH), block = 256 threads (8 wave32)
// Dynamic LDS: sc[16][2048] f32 (128 KB) + colmask[2048] u32 (8 KB, 16 row
// bits per key column) + staged qsig (1 KB)
// ---------------------------------------------------------------------------
__global__ __launch_bounds__(256) void l3k_attn_main(
    const unsigned short* __restrict__ qb,   // [NH][S][D] bf16, pre-scaled
    const unsigned short* __restrict__ kT,   // [KH][S][D] bf16
    const unsigned short* __restrict__ vb,   // [KH][S][D] bf16
    const unsigned* __restrict__ qsig,       // [NH][S][4]
    const unsigned* __restrict__ ksig,       // [KH][S][4]
    const float* __restrict__ thrp,          // [KH]
    const int* __restrict__ topkp,
    const int* __restrict__ wszp,
    float* __restrict__ out)                 // [S][NH*D]
{
  extern __shared__ char smem[];
  float*    sc      = reinterpret_cast<float*>(smem);                    // [BM][S_LEN]
  unsigned* colmask = reinterpret_cast<unsigned*>(smem + 131072);        // [S_LEN]
  unsigned* qsg     = reinterpret_cast<unsigned*>(smem + 131072 + 8192); // [NQH][BM][4]

  const int qblk = blockIdx.x;
  const int kh   = blockIdx.y;
  const int tid  = threadIdx.x;
  const int lane = tid & 31;
  const int wave = tid >> 5;
  const float thr = thrp[kh];
  const int topk = topkp[0];
  const int wsz  = wszp[0];

  // ---- Step 1a: stage the group's query sign bits (4 qh x 16 rows x 4 words)
  {
    const int qh = tid >> 6, r = (tid >> 2) & 15, w = tid & 3;
    qsg[tid] = qsig[((size_t)((kh * NQH + qh) * S_LEN + qblk * BM + r) << 2) + w];
  }
  __syncthreads();

  // ---- Step 1b: column-major LSH mask via popcount; each thread owns 8 keys
  {
    const int c0 = tid * 8;
    unsigned ks0[8][4];
#pragma unroll
    for (int k8 = 0; k8 < 8; ++k8) {
      const unsigned* kp = ksig + ((size_t)(kh * S_LEN + c0 + k8) << 2);
      ks0[k8][0] = kp[0]; ks0[k8][1] = kp[1]; ks0[k8][2] = kp[2]; ks0[k8][3] = kp[3];
    }
    unsigned cm[8] = {0, 0, 0, 0, 0, 0, 0, 0};
    for (int r = 0; r < BM; ++r) {
      int best[8] = {0, 0, 0, 0, 0, 0, 0, 0};
#pragma unroll
      for (int qh = 0; qh < NQH; ++qh) {
        const unsigned q0 = qsg[qh * 64 + r * 4 + 0];
        const unsigned q1 = qsg[qh * 64 + r * 4 + 1];
        const unsigned q2 = qsg[qh * 64 + r * 4 + 2];
        const unsigned q3 = qsg[qh * 64 + r * 4 + 3];
#pragma unroll
        for (int k8 = 0; k8 < 8; ++k8) {
          const int cnt = DHEAD - (__popc(q0 ^ ks0[k8][0]) + __popc(q1 ^ ks0[k8][1]) +
                                   __popc(q2 ^ ks0[k8][2]) + __popc(q3 ^ ks0[k8][3]));
          best[k8] = max(best[k8], cnt);
        }
      }
#pragma unroll
      for (int k8 = 0; k8 < 8; ++k8)
        if ((float)best[k8] > thr) cm[k8] |= (1u << r);
    }
#pragma unroll
    for (int k8 = 0; k8 < 8; ++k8) colmask[c0 + k8] = cm[k8];
  }
  __syncthreads();

  for (int qh = 0; qh < NQH; ++qh) {
    // hoisted Q A-fragments (16x32 bf16 x 4 = full D=128)
    const unsigned short* qhead =
        qb + ((size_t)(kh * NQH + qh) * S_LEN + (size_t)qblk * BM) * DHEAD;
    const int am  = lane & 15;
    const int akb = (lane >> 4) * 8;
    FragB qa[4];
#pragma unroll
    for (int kk = 0; kk < 4; ++kk) {
      const unsigned short* p = qhead + am * DHEAD + kk * 32 + akb;
      qa[kk].q[0] = *reinterpret_cast<const uint4*>(p);
      qa[kk].q[1] = *reinterpret_cast<const uint4*>(p + 16);
    }

    // ---- Step 2: scores = Q K^T via WMMA; branchless mask; f32 stripe in LDS
    for (int kt = wave; kt < S_LEN / 16; kt += 8) {
      if (kt + 8 < S_LEN / 16) {
        __builtin_prefetch(kT + ((size_t)(kh * DHEAD + lane)) * S_LEN + (kt + 8) * 16,
                           0, 1);
      }
      v8f acc = {};
#pragma unroll
      for (int kk = 0; kk < 4; ++kk) {
        FragB bf;
        const unsigned short* p =
            kT + ((size_t)(kh * S_LEN + kt * 16 + (lane & 15))) * DHEAD + kk * 32 + (lane >> 4) * 8;
        bf.q[0] = *reinterpret_cast<const uint4*>(p);
        bf.q[1] = *reinterpret_cast<const uint4*>(p + 16);
        acc = __builtin_amdgcn_wmma_f32_16x16x32_bf16(
            false, qa[kk].v, false, bf.v, (short)0, acc, false, false);
      }
      const int colb = kt * 16 + (lane & 15);
      const unsigned cm = colmask[colb];   // one LDS load per 16x16 tile
#pragma unroll
      for (int j = 0; j < 8; ++j) {
        const int mm = j + ((lane >> 4) << 3);
        const int qi = qblk * BM + mm;
        // pure bitwise predicate -> v_cmp/v_cndmask, no exec-mask branches
        const int keep = (int)(qi >= colb) &
                         ((int)((cm >> mm) & 1u) |
                          (int)((qi - colb) < wsz) | (int)(colb < SINKN));
        sc[mm * S_LEN + colb] = keep ? acc[j] : NEGV;   // Q pre-scaled
      }
    }
    __syncthreads();

    // ---- Step 3: per-row top-k threshold (bisection) + softmax in-place ----
    for (int rr = 0; rr < 2; ++rr) {
      const int r  = wave * 2 + rr;
      const int qi = qblk * BM + r;
      float slt[64];
      float hi = NEGV;
#pragma unroll
      for (int i = 0; i < 64; ++i) {
        const int col = lane + 32 * i;               // bank-conflict-free stride
        const float s = sc[r * S_LEN + col];
        const int win = (int)(col <= qi) &
                        ((int)((qi - col) < wsz) | (int)(col < SINKN));
        const float v = win ? NEGV : s;
        slt[i] = v;
        hi = fmaxf(hi, v);
      }
#pragma unroll
      for (int d = 16; d; d >>= 1) hi = fmaxf(hi, __shfl_xor(hi, d, 32));
      float lo = NEGV;
      for (int it = 0; it < 20; ++it) {
        const float mid = 0.5f * (lo + hi);
        int cnt = 0;
#pragma unroll
        for (int i = 0; i < 64; ++i) cnt += (slt[i] > mid) ? 1 : 0;
#pragma unroll
        for (int d = 16; d; d >>= 1) cnt += __shfl_xor(cnt, d, 32);
        if (cnt > topk) lo = mid; else hi = mid;
      }
      const float tsel = hi;   // count(> tsel) <= top_k

      // replace slt[] with selected raw scores (-3e38 where deselected);
      // exp(-3e38 - mx) == 0, which is exactly the prob we want there.
      float mx = -3.0e38f;
#pragma unroll
      for (int i = 0; i < 64; ++i) {
        const int col = lane + 32 * i;
        const float s = sc[r * S_LEN + col];
        const int win = (int)(col <= qi) &
                        ((int)((qi - col) < wsz) | (int)(col < SINKN));
        const int sel = win | (int)(slt[i] > tsel);
        const float v = sel ? s : -3.0e38f;
        slt[i] = v;
        mx = fmaxf(mx, v);
      }
#pragma unroll
      for (int d = 16; d; d >>= 1) mx = fmaxf(mx, __shfl_xor(mx, d, 32));
      float sum = 0.0f;
#pragma unroll
      for (int i = 0; i < 64; ++i) {
        const float e = __expf(slt[i] - mx);
        slt[i] = e;
        sum += e;
      }
#pragma unroll
      for (int d = 16; d; d >>= 1) sum += __shfl_xor(sum, d, 32);
      const float inv = 1.0f / sum;
#pragma unroll
      for (int i = 0; i < 64; ++i)
        sc[r * S_LEN + lane + 32 * i] = slt[i] * inv;
    }
    __syncthreads();

    // ---- Step 4: out = P V via WMMA (one 16x16 output tile per wave) ----
    {
      const int dbase = wave * 16;
      v8f acc = {};
      const int m2  = lane & 15;
      const int kb2 = (lane >> 4) * 8;
      for (int kc = 0; kc < S_LEN / 32; ++kc) {
        if (kc + 1 < S_LEN / 32) {
          __builtin_prefetch(
              vb + ((size_t)(kh * S_LEN + (kc + 1) * 32 + lane)) * DHEAD + dbase,
              0, 1);
        }
        FragB a;
        const float* p0 = sc + m2 * S_LEN + kc * 32 + kb2;
        const float4 g0 = *reinterpret_cast<const float4*>(p0);
        const float4 g1 = *reinterpret_cast<const float4*>(p0 + 4);
        const float4 g2 = *reinterpret_cast<const float4*>(p0 + 16);
        const float4 g3 = *reinterpret_cast<const float4*>(p0 + 20);
        a.u[0] = pkbf(g0.x, g0.y); a.u[1] = pkbf(g0.z, g0.w);
        a.u[2] = pkbf(g1.x, g1.y); a.u[3] = pkbf(g1.z, g1.w);
        a.u[4] = pkbf(g2.x, g2.y); a.u[5] = pkbf(g2.z, g2.w);
        a.u[6] = pkbf(g3.x, g3.y); a.u[7] = pkbf(g3.z, g3.w);
        FragB b;
        const unsigned short* vp =
            vb + ((size_t)(kh * S_LEN + kc * 32)) * DHEAD + dbase + m2;
#pragma unroll
        for (int i = 0; i < 8; ++i) {
          const int k0 = (i < 4) ? (kb2 + 2 * i) : (16 + kb2 + 2 * (i - 4));
          b.u[i] = (unsigned)vp[k0 * DHEAD] | ((unsigned)vp[(k0 + 1) * DHEAD] << 16);
        }
        acc = __builtin_amdgcn_wmma_f32_16x16x32_bf16(
            false, a.v, false, b.v, (short)0, acc, false, false);
      }
#pragma unroll
      for (int j = 0; j < 8; ++j) {
        const int mm   = j + ((lane >> 4) << 3);
        const int n    = lane & 15;
        const int srow = qblk * BM + mm;
        const int colo = (kh * NQH + qh) * DHEAD + dbase + n;
        out[(size_t)srow * (NH * DHEAD) + colo] = acc[j];
      }
    }
    __syncthreads();
  }
}

// ---------------------------------------------------------------------------
// Host launcher
// ---------------------------------------------------------------------------
extern "C" void kernel_launch(void* const* d_in, const int* in_sizes, int n_in,
                              void* d_out, int out_size, void* d_ws, size_t ws_size,
                              hipStream_t stream) {
  (void)in_sizes; (void)n_in; (void)out_size; (void)ws_size;
  const float* q   = (const float*)d_in[0];
  const float* k   = (const float*)d_in[1];
  const float* v   = (const float*)d_in[2];
  const float* thr = (const float*)d_in[3];
  const int* topk  = (const int*)d_in[4];
  const int* wsz   = (const int*)d_in[5];
  float* out = (float*)d_out;

  char* ws = (char*)d_ws;
  unsigned short* qb  = (unsigned short*)(ws + OFF_QB);
  unsigned short* kTb = (unsigned short*)(ws + OFF_KT);
  unsigned short* vbb = (unsigned short*)(ws + OFF_VB);
  unsigned*       qs  = (unsigned*)(ws + OFF_QS);
  unsigned*       ks  = (unsigned*)(ws + OFF_KS);

  l3k_cvt_bf16<<<(QTOT + 255) / 256, 256, 0, stream>>>(q, qb, QTOT, SCALE);
  l3k_cvt_kT <<<(KTOT + 255) / 256, 256, 0, stream>>>(k, kTb);
  l3k_cvt_bf16<<<(KTOT + 255) / 256, 256, 0, stream>>>(v, vbb, KTOT, 1.0f);
  l3k_sigpack<<<(NH  * S_LEN * 4 + 255) / 256, 256, 0, stream>>>(q, qs, NH  * S_LEN);
  l3k_sigpack<<<(NKH * S_LEN * 4 + 255) / 256, 256, 0, stream>>>(k, ks, NKH * S_LEN);

  const int smem = 131072 + 8192 + 1024;   // 140,288 B (< 320 KB WGP LDS)
  (void)hipFuncSetAttribute(reinterpret_cast<const void*>(l3k_attn_main),
                            hipFuncAttributeMaxDynamicSharedMemorySize, smem);
  l3k_attn_main<<<dim3(S_LEN / BM, NKH), 256, smem, stream>>>(
      qb, kTb, vbb, qs, ks, thr, topk, wsz, out);
}
